// GraphSAGE_31396210934180
// MI455X (gfx1250) — matrix operand order repaired and checked by reading, compiled, not run on hardware
//
#include <hip/hip_runtime.h>
#include <hip/hip_bf16.h>

// GraphSAGE (3x SAGEConv mean-aggr + FC + softmax) for MI455X / gfx1250.
// - edge scatter: wave-per-edge coalesced global_atomic_add_f32 (HBM/atomic bound)
// - GEMMs: v_wmma_f32_16x16x32_f16, f16 operands / f32 accum; weights staged
//   to LDS per block via global_load_async_to_lds_b128 (ASYNCcnt-tracked),
//   fused bias + mean-divide + L2-row-normalize + ReLU epilogue via shuffles
// - final FC 128->40 + softmax fused, padded to N=48 with masked logits

typedef __attribute__((ext_vector_type(16))) _Float16 v16h;
typedef __attribute__((ext_vector_type(8)))  float    v8f;

#define NN   100000
#define NE   1600000
#define HIDD 128
#define OUTC 40

#if __has_builtin(__builtin_amdgcn_global_load_async_to_lds_b128)
#define HAVE_ASYNC_LDS 1
#else
#define HAVE_ASYNC_LDS 0
#endif

// exact pointee type the async-LDS builtin expects: GCC-vector int4
typedef int v4i_vs __attribute__((vector_size(16)));
typedef __attribute__((address_space(1))) v4i_vs glob_v4i;
typedef __attribute__((address_space(3))) v4i_vs lds_v4i;

// Stage n16 16-byte chunks from global into LDS with all 256 threads.
__device__ __forceinline__ void stage_lds(_Float16* dst, const _Float16* src,
                                          int tid, int n16) {
  for (int i = tid; i < n16; i += 256) {
#if HAVE_ASYNC_LDS
    __builtin_amdgcn_global_load_async_to_lds_b128(
        (glob_v4i*)((const char*)src + i * 16),
        (lds_v4i*)((char*)dst + i * 16), 0, 0);
#else
    *(float4*)((char*)dst + i * 16) = *(const float4*)((const char*)src + i * 16);
#endif
  }
}

__device__ __forceinline__ void stage_fence() {
#if HAVE_ASYNC_LDS
#if __has_builtin(__builtin_amdgcn_s_wait_asynccnt)
  __builtin_amdgcn_s_wait_asynccnt(0);
#else
  asm volatile("s_wait_asynccnt 0x0" ::: "memory");
#endif
#endif
  __syncthreads();
}

// ---------------------------------------------------------------- utilities
__global__ void zero_kernel(float* __restrict__ p, int n) {
  int i = blockIdx.x * blockDim.x + threadIdx.x;
  if (i < n) p[i] = 0.0f;
}

// Pre-swizzle weight W[outDim x 128] (row-major, out=n, in=k) into the exact
// WMMA B-fragment lane layout, f16:
//   Wp[((nt*4+ks)*32 + lane)*16 + e]  with  k = ks*32 + (e>>3)*16 + (lane>>4)*8 + (e&7),
//                                           n = nt*16 + (lane&15)
__global__ void wcvt_kernel(const float* __restrict__ W, _Float16* __restrict__ Wp,
                            int outDim, int total) {
  int idx = blockIdx.x * blockDim.x + threadIdx.x;
  if (idx >= total) return;
  int e    = idx & 15;
  int lane = (idx >> 4) & 31;
  int ks   = (idx >> 9) & 3;
  int nt   = idx >> 11;
  int k = ks * 32 + ((e >> 3) << 4) + ((lane >> 4) << 3) + (e & 7);
  int n = nt * 16 + (lane & 15);
  Wp[idx] = (n < outDim) ? (_Float16)W[n * HIDD + k] : (_Float16)0.0f;
}

// ------------------------------------------------------- edge scatter (mean)
__global__ void scatter_kernel(const float* __restrict__ x,
                               const int* __restrict__ src,
                               const int* __restrict__ dst,
                               float* __restrict__ agg,
                               float* __restrict__ cnt) {
  long long tid = (long long)blockIdx.x * blockDim.x + threadIdx.x;
  int edge = (int)(tid >> 5);
  int lane = (int)(tid & 31);
  if (edge >= NE) return;
  int s = src[edge];
  int d = dst[edge];
  float4 v = *(const float4*)(x + s * HIDD + lane * 4);
  float* ad = agg + d * HIDD + lane * 4;
  atomicAdd(ad + 0, v.x);
  atomicAdd(ad + 1, v.y);
  atomicAdd(ad + 2, v.z);
  atomicAdd(ad + 3, v.w);
  if (lane == 0) atomicAdd(cnt + d, 1.0f);
}

// ----------------------------- SAGEConv: out = norm( agg/cnt @ Wl.T + bl + x @ Wr.T )
__global__ __launch_bounds__(256)
void sage_wmma_kernel(const float* __restrict__ x,
                      const float* __restrict__ agg,
                      const float* __restrict__ cnt,
                      const _Float16* __restrict__ Wlp,
                      const float* __restrict__ bl,
                      const _Float16* __restrict__ Wrp,
                      float* __restrict__ out,
                      int do_relu) {
  __shared__ __align__(32) _Float16 lWl[16384];   // 32KB, B fragments of Wl
  __shared__ __align__(32) _Float16 lWr[16384];   // 32KB, B fragments of Wr

  int tid = threadIdx.x;
  stage_lds(lWl, Wlp, tid, 2048);
  stage_lds(lWr, Wrp, tid, 2048);
  stage_fence();                                  // all waves reach barrier

  int wave    = tid >> 5;
  int lane    = tid & 31;
  int rowTile = blockIdx.x * 8 + wave;
  if (rowTile * 16 >= NN) return;                 // wave-uniform: EXEC all-1s
  int row_base = rowTile * 16;
  int m  = lane & 15;
  int kh = lane >> 4;
  int row = row_base + m;

  float inv = 1.0f / fmaxf(cnt[row], 1.0f);
  const float* xr = x   + row * HIDD;
  const float* ar = agg + row * HIDD;

  // A fragments (16x32 f16) for x and mean-aggregated neighbors, K=128 -> 4 steps
  v16h a_x[4], a_g[4];
#pragma unroll
  for (int ks = 0; ks < 4; ++ks) {
    int kbase = ks * 32;
#pragma unroll
    for (int j = 0; j < 8; ++j) {
      int kk = kbase + ((j >> 2) << 4) + (kh << 3) + ((j & 3) << 1);
      float2 px = *(const float2*)(xr + kk);
      float2 pg = *(const float2*)(ar + kk);
      a_x[ks][2 * j]     = (_Float16)px.x;
      a_x[ks][2 * j + 1] = (_Float16)px.y;
      a_g[ks][2 * j]     = (_Float16)(pg.x * inv);
      a_g[ks][2 * j + 1] = (_Float16)(pg.y * inv);
    }
  }

  const v16h* fWl = (const v16h*)lWl;             // one v16h == one B fragment
  const v16h* fWr = (const v16h*)lWr;

  // 8 column tiles of 16 -> full 16x128 output block per wave
  v8f acc[8];
  float bv[8];
#pragma unroll
  for (int nt = 0; nt < 8; ++nt) {
    bv[nt] = bl[nt * 16 + m];                     // C-layout column = lane&15
    v8f c = {};                                   // first WMMA uses inline 0
#pragma unroll
    for (int ks = 0; ks < 4; ++ks) {
      int fi = (nt * 4 + ks) * 32 + lane;
      c = __builtin_amdgcn_wmma_f32_16x16x32_f16(false, a_g[ks], false, fWl[fi],
                                                 (short)0, c, false, false);
      c = __builtin_amdgcn_wmma_f32_16x16x32_f16(false, a_x[ks], false, fWr[fi],
                                                 (short)0, c, false, false);
    }
#pragma unroll
    for (int j = 0; j < 8; ++j) c[j] += bv[nt];   // bias off the WMMA chain
    acc[nt] = c;
  }

  // L2 row norm: VGPR j holds row kh*8+j; reduce over the 16 lanes of a half
  float rs[8];
#pragma unroll
  for (int j = 0; j < 8; ++j) {
    float s = 0.0f;
#pragma unroll
    for (int nt = 0; nt < 8; ++nt) s += acc[nt][j] * acc[nt][j];
    s += __shfl_xor(s, 1);
    s += __shfl_xor(s, 2);
    s += __shfl_xor(s, 4);
    s += __shfl_xor(s, 8);
    rs[j] = 1.0f / fmaxf(sqrtf(s), 1e-12f);
  }

#pragma unroll
  for (int nt = 0; nt < 8; ++nt) {
#pragma unroll
    for (int j = 0; j < 8; ++j) {
      float v = acc[nt][j] * rs[j];
      if (do_relu) v = fmaxf(v, 0.0f);
      out[(row_base + kh * 8 + j) * HIDD + nt * 16 + m] = v;
    }
  }
}

// --------------------------------------- final FC 128->40 + softmax (N padded to 48)
__global__ __launch_bounds__(256)
void fc_softmax_kernel(const float* __restrict__ x,
                       const _Float16* __restrict__ Wp,
                       const float* __restrict__ b,
                       float* __restrict__ out) {
  __shared__ __align__(32) _Float16 lW[6144];     // 12KB, B fragments of Wfc

  int tid = threadIdx.x;
  stage_lds(lW, Wp, tid, 768);
  stage_fence();

  int wave    = tid >> 5;
  int lane    = tid & 31;
  int rowTile = blockIdx.x * 8 + wave;
  if (rowTile * 16 >= NN) return;
  int row_base = rowTile * 16;
  int m  = lane & 15;
  int kh = lane >> 4;
  int row = row_base + m;
  const float* xr = x + row * HIDD;

  v16h a[4];
#pragma unroll
  for (int ks = 0; ks < 4; ++ks) {
    int kbase = ks * 32;
#pragma unroll
    for (int j = 0; j < 8; ++j) {
      int kk = kbase + ((j >> 2) << 4) + (kh << 3) + ((j & 3) << 1);
      float2 p = *(const float2*)(xr + kk);
      a[ks][2 * j]     = (_Float16)p.x;
      a[ks][2 * j + 1] = (_Float16)p.y;
    }
  }

  const v16h* fW = (const v16h*)lW;

  v8f acc[3];
#pragma unroll
  for (int nt = 0; nt < 3; ++nt) {
    int n = nt * 16 + m;
    float bvv = (n < OUTC) ? b[n] : 0.0f;
    v8f c = {};
#pragma unroll
    for (int ks = 0; ks < 4; ++ks) {
      c = __builtin_amdgcn_wmma_f32_16x16x32_f16(false, a[ks], false,
                                                 fW[(nt * 4 + ks) * 32 + lane],
                                                 (short)0, c, false, false);
    }
#pragma unroll
    for (int j = 0; j < 8; ++j) c[j] += bvv;
    acc[nt] = c;
  }

  // mask padded columns, then fused softmax over the 40 valid logits per row
#pragma unroll
  for (int nt = 0; nt < 3; ++nt)
    if (nt * 16 + m >= OUTC) {
#pragma unroll
      for (int j = 0; j < 8; ++j) acc[nt][j] = -1e30f;
    }

  float rsum[8];
#pragma unroll
  for (int j = 0; j < 8; ++j) {
    float mx = -1e30f;
#pragma unroll
    for (int nt = 0; nt < 3; ++nt) mx = fmaxf(mx, acc[nt][j]);
    mx = fmaxf(mx, __shfl_xor(mx, 1));
    mx = fmaxf(mx, __shfl_xor(mx, 2));
    mx = fmaxf(mx, __shfl_xor(mx, 4));
    mx = fmaxf(mx, __shfl_xor(mx, 8));
    float s = 0.0f;
#pragma unroll
    for (int nt = 0; nt < 3; ++nt) {
      acc[nt][j] = __expf(acc[nt][j] - mx);
      s += acc[nt][j];
    }
    s += __shfl_xor(s, 1);
    s += __shfl_xor(s, 2);
    s += __shfl_xor(s, 4);
    s += __shfl_xor(s, 8);
    rsum[j] = 1.0f / s;
  }

#pragma unroll
  for (int nt = 0; nt < 3; ++nt) {
    int n = nt * 16 + m;
    if (n < OUTC) {
#pragma unroll
      for (int j = 0; j < 8; ++j)
        out[(row_base + kh * 8 + j) * OUTC + n] = acc[nt][j] * rsum[j];
    }
  }
}

// ------------------------------------------------------------------- driver
extern "C" void kernel_launch(void* const* d_in, const int* in_sizes, int n_in,
                              void* d_out, int out_size, void* d_ws, size_t ws_size,
                              hipStream_t stream) {
  const float* feat = (const float*)d_in[0];
  const int*   ei   = (const int*)d_in[1];
  const int*   src  = ei;
  const int*   dst  = ei + NE;
  const float* Wl1 = (const float*)d_in[2];  const float* bl1 = (const float*)d_in[3];
  const float* Wr1 = (const float*)d_in[4];
  const float* Wl2 = (const float*)d_in[5];  const float* bl2 = (const float*)d_in[6];
  const float* Wr2 = (const float*)d_in[7];
  const float* Wl3 = (const float*)d_in[8];  const float* bl3 = (const float*)d_in[9];
  const float* Wr3 = (const float*)d_in[10];
  const float* Wfc = (const float*)d_in[11]; const float* bfc = (const float*)d_in[12];

  // workspace carve-out (aligned to 256B)
  char*  ws  = (char*)d_ws;
  size_t off = 0;
  auto carve = [&](size_t bytes) {
    void* p = ws + off;
    off = (off + bytes + 255) & ~(size_t)255;
    return p;
  };
  float*    agg  = (float*)carve((size_t)NN * HIDD * sizeof(float));
  float*    cnt  = (float*)carve((size_t)NN * sizeof(float));
  float*    hA   = (float*)carve((size_t)NN * HIDD * sizeof(float));
  float*    hB   = (float*)carve((size_t)NN * HIDD * sizeof(float));
  _Float16* Wl1p = (_Float16*)carve(16384 * sizeof(_Float16));
  _Float16* Wr1p = (_Float16*)carve(16384 * sizeof(_Float16));
  _Float16* Wl2p = (_Float16*)carve(16384 * sizeof(_Float16));
  _Float16* Wr2p = (_Float16*)carve(16384 * sizeof(_Float16));
  _Float16* Wl3p = (_Float16*)carve(16384 * sizeof(_Float16));
  _Float16* Wr3p = (_Float16*)carve(16384 * sizeof(_Float16));
  _Float16* Wfcp = (_Float16*)carve(6144 * sizeof(_Float16));

  // weight pre-swizzle to WMMA B-fragment layout (f16)
  const int wtot = 16384;                    // 8 nt * 4 ks * 32 lanes * 16
  wcvt_kernel<<<(wtot + 255) / 256, 256, 0, stream>>>(Wl1, Wl1p, HIDD, wtot);
  wcvt_kernel<<<(wtot + 255) / 256, 256, 0, stream>>>(Wr1, Wr1p, HIDD, wtot);
  wcvt_kernel<<<(wtot + 255) / 256, 256, 0, stream>>>(Wl2, Wl2p, HIDD, wtot);
  wcvt_kernel<<<(wtot + 255) / 256, 256, 0, stream>>>(Wr2, Wr2p, HIDD, wtot);
  wcvt_kernel<<<(wtot + 255) / 256, 256, 0, stream>>>(Wl3, Wl3p, HIDD, wtot);
  wcvt_kernel<<<(wtot + 255) / 256, 256, 0, stream>>>(Wr3, Wr3p, HIDD, wtot);
  const int ftot = 6144;                     // 3 nt * 4 ks * 32 * 16
  wcvt_kernel<<<(ftot + 255) / 256, 256, 0, stream>>>(Wfc, Wfcp, OUTC, ftot);

  const int zAgg     = (NN * HIDD + 255) / 256;
  const int zCnt     = (NN + 255) / 256;
  const int scatGrid = (NE * 32 + 255) / 256;        // one wave32 per edge
  const int gemmGrid = ((NN / 16) + 7) / 8;          // 8 row-tiles per block

  // layer 1: feat -> hA (relu)
  zero_kernel<<<zAgg, 256, 0, stream>>>(agg, NN * HIDD);
  zero_kernel<<<zCnt, 256, 0, stream>>>(cnt, NN);
  scatter_kernel<<<scatGrid, 256, 0, stream>>>(feat, src, dst, agg, cnt);
  sage_wmma_kernel<<<gemmGrid, 256, 0, stream>>>(feat, agg, cnt, Wl1p, bl1, Wr1p, hA, 1);

  // layer 2: hA -> hB (relu)
  zero_kernel<<<zAgg, 256, 0, stream>>>(agg, NN * HIDD);
  zero_kernel<<<zCnt, 256, 0, stream>>>(cnt, NN);
  scatter_kernel<<<scatGrid, 256, 0, stream>>>(hA, src, dst, agg, cnt);
  sage_wmma_kernel<<<gemmGrid, 256, 0, stream>>>(hA, agg, cnt, Wl2p, bl2, Wr2p, hB, 1);

  // layer 3: hB -> hA (no relu)
  zero_kernel<<<zAgg, 256, 0, stream>>>(agg, NN * HIDD);
  zero_kernel<<<zCnt, 256, 0, stream>>>(cnt, NN);
  scatter_kernel<<<scatGrid, 256, 0, stream>>>(hB, src, dst, agg, cnt);
  sage_wmma_kernel<<<gemmGrid, 256, 0, stream>>>(hB, agg, cnt, Wl3p, bl3, Wr3p, hA, 0);

  // FC + softmax -> d_out [NN x 40]
  fc_softmax_kernel<<<gemmGrid, 256, 0, stream>>>(hA, Wfcp, bfc, (float*)d_out);
}
